// TransGenerator_80805514707741
// MI455X (gfx1250) — matrix operand order repaired and checked by reading, not compiled, not run on hardware
//
#include <hip/hip_runtime.h>
#include <hip/hip_bf16.h>

// ---------------- problem constants ----------------
constexpr int BB   = 64;
constexpr int W    = 255;
constexpr int S    = 256;      // W + 1 (cls)
constexpr int V    = 16389;
constexpr int D    = 512;
constexpr int NH   = 8;
constexpr int NL   = 8;
constexpr int DH   = 64;       // D / NH
constexpr int FF   = 2048;     // 4*D
constexpr int NLOC = 16384;    // V - 5
constexpr int CLS  = 16388;    // V - 1
constexpr int MROWS = BB * S;  // 16384

typedef __attribute__((ext_vector_type(16))) __bf16 v16bf;
typedef __attribute__((ext_vector_type(8)))  __bf16 v8bf;
typedef __attribute__((ext_vector_type(8)))  float  v8f;

// ---------------------------------------------------------------------------
// Weight packing: f32 [K, rowStride(>=N)] -> bf16 packed in WMMA B-fragment
// order.  For each 32(K)x16(N) tile, lane = (k&16?16:0) + (n&15) holds the 16
// contiguous K-elements  k = khalf*16 .. khalf*16+15  of column n.
// packed[tile*512 + lane*16 + e]  with tile = tk*(N/16)+tn.
// ---------------------------------------------------------------------------
__global__ __launch_bounds__(256) void pack_w_kernel(const float* __restrict__ Wsrc,
                                                     __bf16* __restrict__ P,
                                                     int K, int N, int rowStride) {
    size_t idx = (size_t)blockIdx.x * 256 + threadIdx.x;
    if (idx >= (size_t)K * N) return;
    int n = (int)(idx % N);
    int k = (int)(idx / N);
    float w = Wsrc[(size_t)k * rowStride + n];
    int tk = k >> 5, tn = n >> 4;
    int half = (k >> 4) & 1;
    int e = k & 15;
    int lane = half * 16 + (n & 15);
    size_t dst = ((size_t)(tk * (N >> 4) + tn)) * 512 + (size_t)lane * 16 + e;
    P[dst] = (__bf16)w;
}

// ---------------------------------------------------------------------------
// Embedding + quirky positional encoding:
//   ang = pos / 10000^(2c/D);  even c -> sin, odd c -> cos
// ---------------------------------------------------------------------------
__global__ __launch_bounds__(256) void embed_kernel(const int* __restrict__ x,
                                                    const float* __restrict__ emb,
                                                    float* __restrict__ h) {
    size_t idx = (size_t)blockIdx.x * 256 + threadIdx.x;
    if (idx >= (size_t)MROWS * D) return;
    int d = (int)(idx % D);
    size_t bs = idx / D;
    int s = (int)(bs % S);
    int b = (int)(bs / S);
    int tok = (s < W) ? x[(size_t)b * W + s] : CLS;
    const float LN10000 = 9.210340371976184f;
    float ang = (float)s * expf(-(2.0f * (float)d / (float)D) * LN10000);
    float pe = (d & 1) ? cosf(ang) : sinf(ang);
    h[idx] = emb[(size_t)tok * D + d] + pe;
}

// ---------------------------------------------------------------------------
// LayerNorm over D=512, one block (256 thr) per row, bf16 output (GEMM A feed).
// Row r of input lives at h[(r*rowStride + rowOffset)*D].
// ---------------------------------------------------------------------------
__global__ __launch_bounds__(256) void ln_kernel(const float* __restrict__ h,
                                                 const float* __restrict__ sc,
                                                 const float* __restrict__ bi,
                                                 __bf16* __restrict__ out,
                                                 int rowStride, int rowOffset) {
    __shared__ float red[256];
    int row = blockIdx.x;
    int tid = threadIdx.x;
    const float* xr = h + ((size_t)row * rowStride + rowOffset) * D;
    float x0 = xr[tid], x1 = xr[tid + 256];
    red[tid] = x0 + x1;
    __syncthreads();
    for (int off = 128; off > 0; off >>= 1) {
        if (tid < off) red[tid] += red[tid + off];
        __syncthreads();
    }
    float mean = red[0] * (1.0f / D);
    __syncthreads();
    float d0 = x0 - mean, d1 = x1 - mean;
    red[tid] = d0 * d0 + d1 * d1;
    __syncthreads();
    for (int off = 128; off > 0; off >>= 1) {
        if (tid < off) red[tid] += red[tid + off];
        __syncthreads();
    }
    float rstd = rsqrtf(red[0] * (1.0f / D) + 1e-5f);
    __bf16* orow = out + (size_t)row * D;
    orow[tid]       = (__bf16)(d0 * rstd * sc[tid]       + bi[tid]);
    orow[tid + 256] = (__bf16)(d1 * rstd * sc[tid + 256] + bi[tid + 256]);
}

// ---------------------------------------------------------------------------
// bf16 WMMA GEMM: C[M,N] = A[M,K](bf16,row-major) x Bpacked[K,N](bf16,frag)
// Compile-time epilogue: +bias, +residual, ReLU, f32/bf16 stores, M-guard.
// Block = 256 thr = 8 waves (2Mx4N); wave computes 64x64 (4x4 frags, 16 WMMA
// per 32-wide K step); block covers 128(M) x 256(N).
// Strength-reduced addressing via integer offsets against the kernel-arg base
// pointers keeps loads on the GLOBAL (not FLAT) path.
// ---------------------------------------------------------------------------
template <bool HASBIAS, bool HASRES, bool RELU, bool OUTF, bool OUTB, bool GUARD>
__global__ __launch_bounds__(256) void gemm_bf16_kernel(
    const __bf16* __restrict__ A, const __bf16* __restrict__ Bp,
    const float* __restrict__ bias, const float* __restrict__ resid,
    float* __restrict__ outF, __bf16* __restrict__ outB,
    int M, int N, int K) {
    const int lane = threadIdx.x & 31;
    const int wave = threadIdx.x >> 5;
    const int lm = lane & 15;      // N-index inside fragment / A row select
    const int lh = lane >> 4;      // half select
    const int mBase = blockIdx.y * 128 + (wave & 1) * 64;
    const int nBase = blockIdx.x * 256 + (wave >> 1) * 64;
    const int tilesN = N >> 4;

    v8f acc[4][4];
#pragma unroll
    for (int i = 0; i < 4; ++i)
#pragma unroll
        for (int j = 0; j < 4; ++j) acc[i][j] = (v8f)(0.0f);

    // streaming integer offsets (bases stay kernel-arg pointers => global AS)
    size_t aoff[4];
#pragma unroll
    for (int i = 0; i < 4; ++i) {
        int m = mBase + i * 16 + lm;
        if (GUARD && m > M - 1) m = M - 1;            // clamp (head GEMM, M=64)
        aoff[i] = (size_t)m * K + (size_t)(lh * 16);
    }
    size_t boff[4];
#pragma unroll
    for (int j = 0; j < 4; ++j)
        boff[j] = ((size_t)((nBase + j * 16) >> 4)) * 512 + (size_t)lane * 16;
    const size_t bstep = (size_t)tilesN * 512;        // elements per K-step

    for (int k = 0; k < K; k += 32) {
        v16bf bf[4];
#pragma unroll
        for (int j = 0; j < 4; ++j) {
            bf[j] = *(const v16bf*)(Bp + boff[j]);
            boff[j] += bstep;
        }
        v16bf af[4];
#pragma unroll
        for (int i = 0; i < 4; ++i) {
            v8bf a0 = *(const v8bf*)(A + aoff[i]);
            v8bf a1 = *(const v8bf*)(A + aoff[i] + 8);
            aoff[i] += 32;
            v16bf a;
#pragma unroll
            for (int t = 0; t < 8; ++t) { a[t] = a0[t]; a[t + 8] = a1[t]; }
            af[i] = a;
        }
#pragma unroll
        for (int i = 0; i < 4; ++i)
#pragma unroll
            for (int j = 0; j < 4; ++j)
                acc[i][j] = __builtin_amdgcn_wmma_f32_16x16x32_bf16(
                    false, af[i], false, bf[j], (short)0, acc[i][j], false, false);
    }

    float bv[4];
#pragma unroll
    for (int j = 0; j < 4; ++j)
        bv[j] = HASBIAS ? bias[nBase + j * 16 + lm] : 0.0f;

#pragma unroll
    for (int i = 0; i < 4; ++i) {
        const int mrow0 = mBase + i * 16 + lh * 8;
        size_t rbase = (size_t)mrow0 * N + nBase + lm;  // row-walking offset
#pragma unroll
        for (int r = 0; r < 8; ++r) {
            const int m = mrow0 + r;
            if (!GUARD || m < M) {
#pragma unroll
                for (int j = 0; j < 4; ++j) {
                    float v = acc[i][j][r] + bv[j];
                    if (HASRES) v += resid[rbase + j * 16];
                    if (RELU) v = fmaxf(v, 0.0f);
                    if (OUTF) outF[rbase + j * 16] = v;
                    if (OUTB) outB[rbase + j * 16] = (__bf16)v;
                }
            }
            rbase += N;
        }
    }
}

// ---------------------------------------------------------------------------
// Attention: one block per (b, head). K/V tile in LDS (bf16, 64 KB total),
// one query row per thread, single-pass online softmax. Writes bf16 o.
// qkv layout: [b*S+s, 3D] with q|k|v at col 0|D|2D.
// ---------------------------------------------------------------------------
__global__ __launch_bounds__(256) void attn_kernel(const float* __restrict__ qkv,
                                                   __bf16* __restrict__ obf) {
    __shared__ __bf16 kT[S][DH];
    __shared__ __bf16 vT[S][DH];
    const int bh = blockIdx.x;
    const int b = bh / NH, hh = bh % NH;
    const int tid = threadIdx.x;
    const float* base = qkv + (size_t)b * S * (3 * D);

    for (int i = tid; i < S * DH; i += 256) {
        const int s = i / DH, d = i % DH;
        kT[s][d] = (__bf16)base[(size_t)s * 3 * D + D     + hh * DH + d];
        vT[s][d] = (__bf16)base[(size_t)s * 3 * D + 2 * D + hh * DH + d];
    }
    __syncthreads();

    const int sq = tid;
    float q[DH];
    const float* qrow = base + (size_t)sq * 3 * D + hh * DH;
#pragma unroll
    for (int d = 0; d < DH; ++d) q[d] = qrow[d] * 0.125f;  // 1/sqrt(64)

    float m = -1e30f, l = 0.0f;
    float o[DH];
#pragma unroll
    for (int d = 0; d < DH; ++d) o[d] = 0.0f;

    for (int j = 0; j < S; ++j) {
        float s = 0.0f;
#pragma unroll
        for (int d = 0; d < DH; ++d) s += q[d] * (float)kT[j][d];
        const float mn = fmaxf(m, s);
        const float scl = __expf(m - mn);
        const float p = __expf(s - mn);
        l = l * scl + p;
#pragma unroll
        for (int d = 0; d < DH; ++d) o[d] = o[d] * scl + p * (float)vT[j][d];
        m = mn;
    }
    const float inv = 1.0f / l;
    __bf16* orow = obf + ((size_t)(b * S + sq)) * D + hh * DH;
#pragma unroll
    for (int d = 0; d < DH; ++d) orow[d] = (__bf16)(o[d] * inv);
}

// ---------------------------------------------------------------------------
extern "C" void kernel_launch(void* const* d_in, const int* in_sizes, int n_in,
                              void* d_out, int out_size, void* d_ws, size_t ws_size,
                              hipStream_t stream) {
    const int*   x      = (const int*)d_in[0];
    const float* emb    = (const float*)d_in[1];
    const float* qkv_w  = (const float*)d_in[2];
    const float* qkv_b  = (const float*)d_in[3];
    const float* out_w  = (const float*)d_in[4];
    const float* out_b  = (const float*)d_in[5];
    const float* ln1_s  = (const float*)d_in[6];
    const float* ln1_b  = (const float*)d_in[7];
    const float* ln2_s  = (const float*)d_in[8];
    const float* ln2_b  = (const float*)d_in[9];
    const float* ff1_w  = (const float*)d_in[10];
    const float* ff1_b  = (const float*)d_in[11];
    const float* ff2_w  = (const float*)d_in[12];
    const float* ff2_b  = (const float*)d_in[13];
    const float* fn_s   = (const float*)d_in[14];
    const float* fn_b   = (const float*)d_in[15];
    const float* head_w = (const float*)d_in[16];
    float* out = (float*)d_out;

    // ---- workspace carve-out (256B aligned) ----
    char* base = (char*)d_ws;
    size_t off = 0;
    auto carve = [&](size_t bytes) -> char* {
        char* p = base + off;
        off += (bytes + 255) & ~(size_t)255;
        return p;
    };
    __bf16* Pqkv  = (__bf16*)carve((size_t)NL * D * (3 * D) * 2);
    __bf16* Pout  = (__bf16*)carve((size_t)NL * D * D * 2);
    __bf16* Pff1  = (__bf16*)carve((size_t)NL * D * FF * 2);
    __bf16* Pff2  = (__bf16*)carve((size_t)NL * FF * D * 2);
    __bf16* Phead = (__bf16*)carve((size_t)D * NLOC * 2);
    float*  h     = (float*)carve((size_t)MROWS * D * 4);
    __bf16* x2bf  = (__bf16*)carve((size_t)MROWS * D * 2);
    float*  qkvF  = (float*)carve((size_t)MROWS * (3 * D) * 4);
    __bf16* obf   = (__bf16*)carve((size_t)MROWS * D * 2);
    __bf16* ff1bf = (__bf16*)carve((size_t)MROWS * FF * 2);
    __bf16* fnbf  = (__bf16*)carve((size_t)BB * D * 2);
    (void)ws_size; (void)n_in; (void)in_sizes; (void)out_size;

    auto packGrid = [](size_t elems) { return dim3((unsigned)((elems + 255) / 256)); };

    // ---- pack weights to WMMA fragment layout (bf16) ----
    for (int l = 0; l < NL; ++l) {
        pack_w_kernel<<<packGrid((size_t)D * 3 * D), 256, 0, stream>>>(
            qkv_w + (size_t)l * D * 3 * D, Pqkv + (size_t)l * D * 3 * D, D, 3 * D, 3 * D);
        pack_w_kernel<<<packGrid((size_t)D * D), 256, 0, stream>>>(
            out_w + (size_t)l * D * D, Pout + (size_t)l * D * D, D, D, D);
        pack_w_kernel<<<packGrid((size_t)D * FF), 256, 0, stream>>>(
            ff1_w + (size_t)l * D * FF, Pff1 + (size_t)l * D * FF, D, FF, FF);
        pack_w_kernel<<<packGrid((size_t)FF * D), 256, 0, stream>>>(
            ff2_w + (size_t)l * FF * D, Pff2 + (size_t)l * FF * D, FF, D, D);
    }
    pack_w_kernel<<<packGrid((size_t)D * NLOC), 256, 0, stream>>>(
        head_w, Phead, D, NLOC, V);

    // ---- embedding + positional ----
    embed_kernel<<<packGrid((size_t)MROWS * D), 256, 0, stream>>>(x, emb, h);

    // ---- transformer layers ----
    for (int l = 0; l < NL; ++l) {
        ln_kernel<<<MROWS, 256, 0, stream>>>(h, ln1_s + (size_t)l * D,
                                             ln1_b + (size_t)l * D, x2bf, 1, 0);
        // qkv: bias, f32 out
        gemm_bf16_kernel<true, false, false, true, false, false>
            <<<dim3((3 * D) / 256, MROWS / 128), 256, 0, stream>>>(
                x2bf, Pqkv + (size_t)l * D * 3 * D, qkv_b + (size_t)l * 3 * D,
                nullptr, qkvF, nullptr, MROWS, 3 * D, D);
        attn_kernel<<<BB * NH, 256, 0, stream>>>(qkvF, obf);
        // out-proj: bias + residual(h), f32 out -> h
        gemm_bf16_kernel<true, true, false, true, false, false>
            <<<dim3(D / 256, MROWS / 128), 256, 0, stream>>>(
                obf, Pout + (size_t)l * D * D, out_b + (size_t)l * D,
                h, h, nullptr, MROWS, D, D);
        ln_kernel<<<MROWS, 256, 0, stream>>>(h, ln2_s + (size_t)l * D,
                                             ln2_b + (size_t)l * D, x2bf, 1, 0);
        // ff1: bias + ReLU, bf16 out
        gemm_bf16_kernel<true, false, true, false, true, false>
            <<<dim3(FF / 256, MROWS / 128), 256, 0, stream>>>(
                x2bf, Pff1 + (size_t)l * D * FF, ff1_b + (size_t)l * FF,
                nullptr, nullptr, ff1bf, MROWS, FF, D);
        // ff2: bias + residual(h), f32 out -> h
        gemm_bf16_kernel<true, true, false, true, false, false>
            <<<dim3(D / 256, MROWS / 128), 256, 0, stream>>>(
                ff1bf, Pff2 + (size_t)l * FF * D, ff2_b + (size_t)l * D,
                h, h, nullptr, MROWS, D, FF);
    }

    // ---- final LN on last position only, then head projection ----
    ln_kernel<<<BB, 256, 0, stream>>>(h, fn_s, fn_b, fnbf, S, S - 1);
    // head: no bias, guarded M=64, f32 out -> d_out
    gemm_bf16_kernel<false, false, false, true, false, true>
        <<<dim3(NLOC / 256, 1), 256, 0, stream>>>(
            fnbf, Phead, nullptr, nullptr, out, nullptr, BB, NLOC, D);
}
